// GroupedQueryAttention_58420145160441
// MI455X (gfx1250) — matrix operand order repaired and checked by reading, not compiled, not run on hardware
//
#include <hip/hip_runtime.h>
#include <stdint.h>

// ---------------- types ----------------
typedef __attribute__((ext_vector_type(16))) __bf16 v16bf;
typedef __attribute__((ext_vector_type(8)))  __bf16 v8bf;
typedef __attribute__((ext_vector_type(8)))  float  v8f;
typedef __attribute__((ext_vector_type(4)))  unsigned int v4u;
typedef __attribute__((ext_vector_type(8)))  unsigned int v8u;

#define DEV static __device__ __forceinline__

// ---------------- problem constants ----------------
constexpr int B_  = 2;
constexpr int T_  = 2048;
constexpr int DIN = 2048;
constexpr int H_  = 16;
constexpr int G_  = 4;
constexpr int HD_ = 128;
constexpr int M_  = B_ * T_;                 // 4096 token rows
constexpr int NQ  = H_ * HD_;                // 2048
constexpr int NK  = G_ * HD_;                // 512
constexpr int NQKV = NQ + 2 * NK;            // 3072

// ---------------- fragment helpers ----------------
DEV v16bf mk_frag(v4u lo, v4u hi) {
    v8u u = __builtin_shufflevector(lo, hi, 0, 1, 2, 3, 4, 5, 6, 7);
    return __builtin_bit_cast(v16bf, u);
}
DEV v4u ld4g(const __bf16* p) { return *(const v4u*)p; }   // 16B load (global or LDS)

DEV v8f wmma_bf16(v16bf a, v16bf b, v8f c) {
    return __builtin_amdgcn_wmma_f32_16x16x32_bf16(
        /*neg_a=*/false, a, /*neg_b=*/false, b,
        /*c_mod=*/(short)0, c, /*reuse_a=*/false, /*reuse_b=*/false);
}

// ---------------- CDNA5 async global->LDS copy (ASYNCcnt path) ----------------
// Each active lane copies 16B from its global address to its LDS address.
// LDS byte offset = low 32 bits of the flat pointer to __shared__ memory
// (flat aperture keeps the wave-relative LDS offset in addr[31:0]).
DEV unsigned lds_off(const void* p) { return (unsigned)(uintptr_t)p; }
DEV void async_ld16(unsigned lds_addr, const __bf16* gptr) {
    asm volatile("global_load_async_to_lds_b128 %0, %1, off"
                 :: "v"(lds_addr), "v"((unsigned long long)(uintptr_t)gptr)
                 : "memory");
}
DEV void wait_async0() { asm volatile("s_wait_asynccnt 0x0" ::: "memory"); }
DEV void fence_ds()    { asm volatile("s_wait_dscnt 0x0"    ::: "memory"); }

// ---------------- kernel 1: fp32 -> bf16 pack ----------------
__global__ void cvt_f32_bf16(const float* __restrict__ s, __bf16* __restrict__ d, int n) {
    int i = blockIdx.x * blockDim.x + threadIdx.x;
    int stride = gridDim.x * blockDim.x;
    for (; i < n; i += stride) d[i] = (__bf16)s[i];
}

// ---------------- kernel 2/5: C[M][N] = A[M][K](bf16) @ W[N][K]^T (bf16) -> f32 ----------------
// Block: 256 threads = 8 waves. Block tile 128x128, K-step 32, double-buffered async LDS.
// Wave (wm, wn): wm = w&3 -> 32-row strip, wn = w>>2 -> 64-col strip. 8 WMMA / wave / K-step.
__global__ void __launch_bounds__(256)
gemm_bf16_nt(const __bf16* __restrict__ A, const __bf16* __restrict__ W,
             float* __restrict__ C, int M, int N, int K, int mtiles) {
    __shared__ __align__(16) __bf16 As[2][128 * 32];
    __shared__ __align__(16) __bf16 Bs[2][128 * 32];
    const int m0 = (blockIdx.x % mtiles) * 128;
    const int n0 = (blockIdx.x / mtiles) * 128;
    const int tid  = threadIdx.x;
    const int lane = tid & 31;
    const int w    = tid >> 5;
    const int wm   = w & 3;       // 0..3 -> 32-row strip
    const int wn   = w >> 2;      // 0..1 -> 64-col strip
    const int lrow = lane & 15;
    const int hi   = lane >> 4;
    v8f acc[2][4] = {};

    auto stage = [&](int buf, int k0) {
        #pragma unroll
        for (int i = 0; i < 2; ++i) {
            int c   = tid * 2 + i;          // 0..511
            int row = c >> 2;               // 0..127
            int off = (c & 3) * 8;          // halves
            async_ld16(lds_off(&As[buf][row * 32 + off]),
                       &A[(size_t)(m0 + row) * K + k0 + off]);
            async_ld16(lds_off(&Bs[buf][row * 32 + off]),
                       &W[(size_t)(n0 + row) * K + k0 + off]);
        }
    };

    stage(0, 0);
    wait_async0();
    __syncthreads();

    int buf = 0;
    for (int k0 = 0; k0 < K; k0 += 32) {
        if (k0 + 32 < K) stage(buf ^ 1, k0 + 32);   // prefetch next K-slice

        v16bf af[2];
        #pragma unroll
        for (int mi = 0; mi < 2; ++mi) {
            int arow = wm * 32 + mi * 16 + lrow;
            af[mi] = mk_frag(ld4g(&As[buf][arow * 32 + hi * 8]),
                             ld4g(&As[buf][arow * 32 + 16 + hi * 8]));
        }
        #pragma unroll
        for (int ni = 0; ni < 4; ++ni) {
            int brow = wn * 64 + ni * 16 + lrow;
            v16bf bf = mk_frag(ld4g(&Bs[buf][brow * 32 + hi * 8]),
                               ld4g(&Bs[buf][brow * 32 + 16 + hi * 8]));
            acc[0][ni] = wmma_bf16(af[0], bf, acc[0][ni]);
            acc[1][ni] = wmma_bf16(af[1], bf, acc[1][ni]);
        }

        wait_async0();
        __syncthreads();
        buf ^= 1;
    }

    #pragma unroll
    for (int mi = 0; mi < 2; ++mi)
        #pragma unroll
        for (int ni = 0; ni < 4; ++ni)
            #pragma unroll
            for (int r = 0; r < 8; ++r) {
                int mm = m0 + wm * 32 + mi * 16 + r + hi * 8;
                int nn = n0 + wn * 64 + ni * 16 + lrow;
                C[(size_t)mm * N + nn] = acc[mi][ni][r];
            }
}

// ---------------- kernel 3: RMSNorm + RoPE + bf16 repack ----------------
// One wave per 128-wide head row. Rows per (b,t): 16 q + 4 k + 4 v = 24.
__global__ void __launch_bounds__(128)
norm_rope(const float* __restrict__ qkv,
          const float* __restrict__ cosp, const float* __restrict__ sinp,
          const float* __restrict__ qw, const float* __restrict__ kw,
          __bf16* __restrict__ qb, __bf16* __restrict__ kb, __bf16* __restrict__ vb) {
    const int w    = blockIdx.x * 4 + (threadIdx.x >> 5);
    const int lane = threadIdx.x & 31;
    const int r24  = w % 24;
    const int bt   = w / 24;
    const int b    = bt / T_;
    const int t    = bt % T_;

    const float* src; __bf16* dst; const float* nw = nullptr;
    if (r24 < H_) {
        src = qkv + (size_t)bt * NQKV + r24 * HD_;
        dst = qb + ((size_t)(b * H_ + r24) * T_ + t) * HD_;
        nw  = qw;
    } else if (r24 < H_ + G_) {
        int g = r24 - H_;
        src = qkv + (size_t)bt * NQKV + NQ + g * HD_;
        dst = kb + ((size_t)(b * G_ + g) * T_ + t) * HD_;
        nw  = kw;
    } else {
        int g = r24 - H_ - G_;
        src = qkv + (size_t)bt * NQKV + NQ + NK + g * HD_;
        dst = vb + ((size_t)(b * G_ + g) * T_ + t) * HD_;
    }

    float u0 = src[lane], u1 = src[lane + 32], u2 = src[lane + 64], u3 = src[lane + 96];
    if (nw) {
        float ss = u0 * u0 + u1 * u1 + u2 * u2 + u3 * u3;
        #pragma unroll
        for (int m = 16; m >= 1; m >>= 1) ss += __shfl_xor(ss, m, 32);
        float rn = rsqrtf(ss * (1.0f / HD_) + 1e-6f);
        u0 *= rn * nw[lane];      u1 *= rn * nw[lane + 32];
        u2 *= rn * nw[lane + 64]; u3 *= rn * nw[lane + 96];
        const float* cr = cosp + (size_t)t * HD_;
        const float* sr = sinp + (size_t)t * HD_;
        float o0 = u0 * cr[lane]      - u2 * sr[lane];        // d < 64: u*cos - u[d+64]*sin
        float o1 = u1 * cr[lane + 32] - u3 * sr[lane + 32];
        float o2 = u2 * cr[lane + 64] + u0 * sr[lane + 64];   // d >= 64: u*cos + u[d-64]*sin
        float o3 = u3 * cr[lane + 96] + u1 * sr[lane + 96];
        u0 = o0; u1 = o1; u2 = o2; u3 = o3;
    }
    dst[lane]      = (__bf16)u0; dst[lane + 32] = (__bf16)u1;
    dst[lane + 64] = (__bf16)u2; dst[lane + 96] = (__bf16)u3;
}

// ---------------- kernel 4: causal flash attention ----------------
// Block: 128 threads = 4 waves = 64 consecutive queries of one head.
// K/V chunks (32 keys) staged once per block and shared by all 4 waves.
__global__ void __launch_bounds__(128)
attn_fwd(const __bf16* __restrict__ qb, const __bf16* __restrict__ kb,
         const __bf16* __restrict__ vb, __bf16* __restrict__ ctxb) {
    __shared__ __align__(16) __bf16 Kl[32 * 128];     // K chunk, row-major [key][hd]
    __shared__ __align__(16) __bf16 Vt[128 * 32];     // V chunk, transposed [hd][key]
    __shared__ __align__(16) __bf16 Pl[4][16 * 32];   // per-wave probabilities [qrow][key]

    const int qt64 = blockIdx.x % (T_ / 64);
    const int bh   = blockIdx.x / (T_ / 64);
    const int b    = bh / H_;
    const int h    = bh % H_;
    const int g    = h / (H_ / G_);
    const int tid  = threadIdx.x;
    const int lane = tid & 31;
    const int w    = tid >> 5;
    const int lrow = lane & 15;
    const int hi   = lane >> 4;
    const int q0   = qt64 * 64 + w * 16;      // this wave's query tile base

    // Q tile fragments (A-matrix, 4 K-slices of 32 over head-dim 128)
    const __bf16* qrow = qb + ((size_t)(b * H_ + h) * T_ + q0 + lrow) * HD_;
    v16bf qf[4];
    #pragma unroll
    for (int c = 0; c < 4; ++c)
        qf[c] = mk_frag(ld4g(qrow + c * 32 + hi * 8),
                        ld4g(qrow + c * 32 + 16 + hi * 8));

    const __bf16* kbase = kb + (size_t)(b * G_ + g) * T_ * HD_;
    const __bf16* vbase = vb + (size_t)(b * G_ + g) * T_ * HD_;

    v8f ctx[8] = {};
    float mrow[8], lrun[8];
    #pragma unroll
    for (int r = 0; r < 8; ++r) { mrow[r] = -1e30f; lrun[r] = 0.0f; }
    const float scale = 0.08838834764831845f;   // 1/sqrt(128)

    const int jmax = qt64 * 64 + 63;            // block-level causal bound
    for (int j0 = 0; j0 <= jmax; j0 += 32) {
        // --- stage K chunk (32x128) via async global->LDS: 512 x 16B, 4 per thread ---
        #pragma unroll
        for (int i = 0; i < 4; ++i) {
            int c   = tid * 4 + i;      // 0..511
            int row = c >> 4;           // 0..31
            int off = (c & 15) * 8;     // halves
            int jr  = j0 + row; if (jr > T_ - 1) jr = T_ - 1;
            async_ld16(lds_off(&Kl[row * 128 + off]), kbase + (size_t)jr * HD_ + off);
        }
        // --- stage V chunk transposed (manual: needs lane-level scatter) ---
        {
            int row = tid >> 2;                        // 4 threads per key row
            int jr  = j0 + row; if (jr > T_ - 1) jr = T_ - 1;
            const __bf16* vr = vbase + (size_t)jr * HD_;
            #pragma unroll
            for (int dd = 0; dd < 4; ++dd) {
                int d8 = (tid & 3) * 4 + dd;           // 16B chunk of the row
                v8bf vv = __builtin_bit_cast(v8bf, ld4g(vr + d8 * 8));
                #pragma unroll
                for (int i = 0; i < 8; ++i) Vt[(d8 * 8 + i) * 32 + row] = vv[i];
            }
        }
        wait_async0();
        __syncthreads();

        if (j0 <= q0 + 15) {   // wave-uniform causal gate
            // scores S[16q][32k]: two 16x16 accumulators, K-dim 128 in 4 WMMA steps each
            v8f s0 = {}, s1 = {};
            #pragma unroll
            for (int c = 0; c < 4; ++c) {
                v16bf bk0 = mk_frag(ld4g(&Kl[lrow * 128 + c * 32 + hi * 8]),
                                    ld4g(&Kl[lrow * 128 + c * 32 + 16 + hi * 8]));
                v16bf bk1 = mk_frag(ld4g(&Kl[(16 + lrow) * 128 + c * 32 + hi * 8]),
                                    ld4g(&Kl[(16 + lrow) * 128 + c * 32 + 16 + hi * 8]));
                s0 = wmma_bf16(qf[c], bk0, s0);
                s1 = wmma_bf16(qf[c], bk1, s1);
            }

            // online softmax per query row (rows split across lane halves)
            #pragma unroll
            for (int r = 0; r < 8; ++r) {
                int qi = q0 + r + hi * 8;
                int kj = j0 + lrow;
                float a0 = s0[r] * scale, a1 = s1[r] * scale;
                if (kj > qi)      a0 = -1e30f;
                if (kj + 16 > qi) a1 = -1e30f;
                float mx = fmaxf(a0, a1);
                #pragma unroll
                for (int m = 8; m >= 1; m >>= 1) mx = fmaxf(mx, __shfl_xor(mx, m, 16));
                float mn = fmaxf(mrow[r], mx);
                float p0 = __expf(a0 - mn), p1 = __expf(a1 - mn);
                float alpha = __expf(mrow[r] - mn);
                mrow[r] = mn;
                float ls = p0 + p1;
                #pragma unroll
                for (int m = 8; m >= 1; m >>= 1) ls += __shfl_xor(ls, m, 16);
                lrun[r] = lrun[r] * alpha + ls;
                #pragma unroll
                for (int s = 0; s < 8; ++s) ctx[s][r] *= alpha;
                Pl[w][(r + hi * 8) * 32 + lrow]      = (__bf16)p0;
                Pl[w][(r + hi * 8) * 32 + 16 + lrow] = (__bf16)p1;
            }
            fence_ds();   // same-wave DS in-order: make P stores visible to frag loads

            // ctx += P(16x32) @ V(32x128): P as A-matrix, Vt columns as B-matrix
            v16bf pf = mk_frag(ld4g(&Pl[w][lrow * 32 + hi * 8]),
                               ld4g(&Pl[w][lrow * 32 + 16 + hi * 8]));
            #pragma unroll
            for (int s = 0; s < 8; ++s) {
                const int col = s * 16 + lrow;
                v16bf vf = mk_frag(ld4g(&Vt[col * 32 + hi * 8]),
                                   ld4g(&Vt[col * 32 + 16 + hi * 8]));
                ctx[s] = wmma_bf16(pf, vf, ctx[s]);
            }
        }
        __syncthreads();
    }

    // normalize and write ctx bf16 in [token][h*HD] layout for the output GEMM
    #pragma unroll
    for (int r = 0; r < 8; ++r) {
        float invl = 1.0f / lrun[r];
        int   mm   = q0 + r + hi * 8;
        __bf16* orow = ctxb + ((size_t)b * T_ + mm) * (size_t)NQ + h * HD_;
        #pragma unroll
        for (int s = 0; s < 8; ++s)
            orow[s * 16 + lrow] = (__bf16)(ctx[s][r] * invl);
    }
}

// ---------------- launcher ----------------
extern "C" void kernel_launch(void* const* d_in, const int* in_sizes, int n_in,
                              void* d_out, int out_size, void* d_ws, size_t ws_size,
                              hipStream_t stream) {
    const float* x    = (const float*)d_in[0];
    const float* cosp = (const float*)d_in[1];
    const float* sinp = (const float*)d_in[2];
    const float* wq   = (const float*)d_in[3];
    const float* wk   = (const float*)d_in[4];
    const float* wv   = (const float*)d_in[5];
    const float* wo   = (const float*)d_in[6];
    const float* qnw  = (const float*)d_in[7];
    const float* knw  = (const float*)d_in[8];

    char* ws = (char*)d_ws;
    size_t off = 0;
    __bf16* xb    = (__bf16*)(ws + off); off += (size_t)M_ * DIN * 2;          // 16 MiB
    __bf16* wqkv  = (__bf16*)(ws + off); off += (size_t)NQKV * DIN * 2;        // 12 MiB
    __bf16* wob   = (__bf16*)(ws + off); off += (size_t)DIN * NQ * 2;          //  8 MiB
    float*  qkv   = (float*) (ws + off); off += (size_t)M_ * NQKV * 4;         // 48 MiB
    __bf16* qb    = (__bf16*)(ws + off); off += (size_t)B_ * H_ * T_ * HD_ * 2;
    __bf16* kb    = (__bf16*)(ws + off); off += (size_t)B_ * G_ * T_ * HD_ * 2;
    __bf16* vb    = (__bf16*)(ws + off); off += (size_t)B_ * G_ * T_ * HD_ * 2;
    __bf16* ctxb  = (__bf16*)(ws + off); off += (size_t)M_ * NQ * 2;

    auto cvt_blocks = [](int n) { int b = (n + 255) / 256; return b > 4096 ? 4096 : b; };

    // 1) pack to bf16 (Wq/Wk/Wv concatenated row-wise: features match qkv column layout)
    cvt_f32_bf16<<<cvt_blocks(M_ * DIN), 256, 0, stream>>>(x, xb, M_ * DIN);
    cvt_f32_bf16<<<cvt_blocks(NQ * DIN), 256, 0, stream>>>(wq, wqkv, NQ * DIN);
    cvt_f32_bf16<<<cvt_blocks(NK * DIN), 256, 0, stream>>>(wk, wqkv + (size_t)NQ * DIN, NK * DIN);
    cvt_f32_bf16<<<cvt_blocks(NK * DIN), 256, 0, stream>>>(wv, wqkv + (size_t)(NQ + NK) * DIN, NK * DIN);
    cvt_f32_bf16<<<cvt_blocks(DIN * NQ), 256, 0, stream>>>(wo, wob, DIN * NQ);

    // 2) QKV projection: [4096 x 3072] = xb @ wqkv^T
    {
        int mtiles = M_ / 128, ntiles = NQKV / 128;
        gemm_bf16_nt<<<dim3(mtiles * ntiles), 256, 0, stream>>>(xb, wqkv, qkv, M_, NQKV, DIN, mtiles);
    }

    // 3) RMSNorm + RoPE + repack bf16
    norm_rope<<<dim3(B_ * T_ * 24 / 4), 128, 0, stream>>>(qkv, cosp, sinp, qnw, knw, qb, kb, vb);

    // 4) causal flash attention (4 waves / block share K/V staging)
    attn_fwd<<<dim3(B_ * H_ * (T_ / 64)), 128, 0, stream>>>(qb, kb, vb, ctxb);

    // 5) output projection: [4096 x 2048] = ctxb @ wob^T -> d_out (fp32)
    {
        int mtiles = M_ / 128, ntiles = NQ / 128;
        gemm_bf16_nt<<<dim3(mtiles * ntiles), 256, 0, stream>>>(ctxb, wob, (float*)d_out, M_, NQ, DIN, mtiles);
    }
}
